// GPT_21655225106767
// MI455X (gfx1250) — matrix operand order repaired and checked by reading, not compiled, not run on hardware
//
#include <hip/hip_runtime.h>
#include <hip/hip_bf16.h>

typedef _Float16 f16;
typedef __attribute__((ext_vector_type(16))) _Float16 v16h;
typedef __attribute__((ext_vector_type(8)))  _Float16 v8h;
typedef __attribute__((ext_vector_type(8)))  float    v8f;

// Model constants (fixed by the reference)
static constexpr int BATCH = 128;
static constexpr int SEQ   = 256;
static constexpr int M     = BATCH * SEQ;   // 32768 rows
static constexpr int DMODEL= 384;
static constexpr int NHEAD = 6;
static constexpr int HD    = 64;
static constexpr int NLAYER= 6;
static constexpr int FF    = 1536;
static constexpr int VOCAB = 95;
static constexpr int VPAD  = 128;

// ---------------------------------------------------------------------------
// Embedding: x[row, d] = tok_emb[idx[row], d] + pos_emb[row % T, d]
// ---------------------------------------------------------------------------
__global__ void embed_kernel(const int* __restrict__ idx,
                             const float* __restrict__ tok,
                             const float* __restrict__ pos,
                             float* __restrict__ X) {
    size_t i = (size_t)blockIdx.x * 256 + threadIdx.x;   // over M*D
    int row = (int)(i / DMODEL);
    int d   = (int)(i % DMODEL);
    int t   = row & (SEQ - 1);
    X[i] = tok[(size_t)idx[row] * DMODEL + d] + pos[(size_t)t * DMODEL + d];
}

// ---------------------------------------------------------------------------
// Weight convert+transpose: W[K,N] f32 -> out[Npad,K] f16 (zero-padded rows)
// ---------------------------------------------------------------------------
__global__ void wconv_kernel(const float* __restrict__ W, f16* __restrict__ out,
                             int K, int N, int Npad) {
    size_t i = (size_t)blockIdx.x * 256 + threadIdx.x;
    if (i >= (size_t)Npad * K) return;
    int n = (int)(i / K);
    int k = (int)(i % K);
    out[i] = (n < N) ? (f16)W[(size_t)k * N + n] : (f16)0.f;
}

// ---------------------------------------------------------------------------
// f32 -> f16 elementwise (for attention output feeding the O-proj GEMM)
// ---------------------------------------------------------------------------
__global__ void cvt_kernel(const float* __restrict__ in, f16* __restrict__ out) {
    size_t i = (size_t)blockIdx.x * 256 + threadIdx.x;
    out[i] = (f16)in[i];
}

// ---------------------------------------------------------------------------
// LayerNorm over D=384, one wave32 per row, f16 output (GEMM input format)
// ---------------------------------------------------------------------------
__global__ __launch_bounds__(256) void ln_kernel(const float* __restrict__ X,
                                                 const float* __restrict__ s,
                                                 const float* __restrict__ b,
                                                 f16* __restrict__ out) {
    int wave = threadIdx.x >> 5;
    int lane = threadIdx.x & 31;
    int row  = blockIdx.x * 8 + wave;
    const float* xr = X + (size_t)row * DMODEL;

    float v[12];
    float sum = 0.f, sq = 0.f;
#pragma unroll
    for (int i = 0; i < 12; ++i) {
        v[i] = xr[lane + 32 * i];
        sum += v[i];
        sq  += v[i] * v[i];
    }
#pragma unroll
    for (int off = 16; off > 0; off >>= 1) {
        sum += __shfl_xor(sum, off, 32);
        sq  += __shfl_xor(sq,  off, 32);
    }
    const float inv_d = 1.f / (float)DMODEL;
    float mean = sum * inv_d;
    float var  = sq * inv_d - mean * mean;
    float rstd = rsqrtf(var + 1e-5f);

    f16* orow = out + (size_t)row * DMODEL;
#pragma unroll
    for (int i = 0; i < 12; ++i) {
        int c = lane + 32 * i;
        orow[c] = (f16)((v[i] - mean) * rstd * s[c] + b[c]);
    }
}

// ---------------------------------------------------------------------------
// WMMA GEMM: C[M,N] = A[M,K](f16) * Bt[Npad,K](f16, transposed) (+bias)(+res)
// Block = 8 waves; each wave computes a 32x64 strip: 2 A frags x 4 B frags ->
// 8 x v_wmma_f32_16x16x32_f16 per K-step of 32. K loop unrolled by 2 with
// explicit double-buffered fragments so loads for step k+32 are in flight
// while the WMMAs for step k execute (K is always a multiple of 64 here).
// Epilogue fuses bias, residual add, ReLU, f16 store.
// ---------------------------------------------------------------------------
__global__ __launch_bounds__(256) void gemm_wmma_kernel(
    const f16* __restrict__ A, const f16* __restrict__ Bt,
    const float* __restrict__ bias, const float* __restrict__ res,
    float* __restrict__ outF, f16* __restrict__ outH,
    int K, int N, int relu) {
    const int lane  = threadIdx.x & 31;
    const int wave  = threadIdx.x >> 5;
    const int laneM = lane & 15;
    const int hi    = lane >> 4;            // 0: lanes 0-15, 1: lanes 16-31

    const int rowBase = blockIdx.y * 256 + wave * 32;
    const int colBase = blockIdx.x * 64;

    v8f zero = {0.f, 0.f, 0.f, 0.f, 0.f, 0.f, 0.f, 0.f};
    v8f acc[2][4] = {{zero, zero, zero, zero}, {zero, zero, zero, zero}};

    // A fragment element addresses (contiguous k order, same as B):
    //   lane<16 : K = k0+0..15 of row  laneM
    //   lane>=16: K = k0+16..31 of row laneM
    const f16* aptr = A + (size_t)(rowBase + laneM) * K + hi * 16;
    // B fragment (32x16, K-major per column, column = colBase+16n+laneM):
    //   lane<16 holds K = k0+0..15, lane>=16 holds K = k0+16..31
    const f16* bptr = Bt + (size_t)(colBase + laneM) * K + hi * 16;

    auto loadA = [&](v16h (&ab)[2], int k0) {
#pragma unroll
        for (int mi = 0; mi < 2; ++mi) {
            union { v16h v; v8h h[2]; } t;
            t.h[0] = *(const v8h*)(aptr + (size_t)mi * 16 * K + k0);
            t.h[1] = *(const v8h*)(aptr + (size_t)mi * 16 * K + k0 + 8);
            ab[mi] = t.v;
        }
    };
    auto loadB = [&](v16h (&bb)[4], int k0) {
#pragma unroll
        for (int n = 0; n < 4; ++n)
            bb[n] = *(const v16h*)(bptr + (size_t)n * 16 * K + k0);
    };
    auto mma = [&](v16h (&ab)[2], v16h (&bb)[4]) {
#pragma unroll
        for (int n = 0; n < 4; ++n)
#pragma unroll
            for (int mi = 0; mi < 2; ++mi)
                acc[mi][n] = __builtin_amdgcn_wmma_f32_16x16x32_f16(
                    false, ab[mi], false, bb[n], (short)0, acc[mi][n],
                    false, false);
    };

    v16h a0[2], a1[2], b0[4], b1[4];
    loadA(a0, 0);
    loadB(b0, 0);

    for (int k0 = 0; k0 < K; k0 += 64) {
        // stage k0+32 while consuming k0
        loadA(a1, k0 + 32);
        loadB(b1, k0 + 32);
        __builtin_prefetch((const void*)(aptr + k0 + 64), 0, 1);
        mma(a0, b0);
        // stage k0+64 (if any) while consuming k0+32
        if (k0 + 64 < K) {
            loadA(a0, k0 + 64);
            loadB(b0, k0 + 64);
        }
        mma(a1, b1);
    }

    // Epilogue. C VGPR r holds row (rowBase + mi*16 + r + hi*8),
    // column colBase + 16n + laneM.
#pragma unroll
    for (int mi = 0; mi < 2; ++mi) {
#pragma unroll
        for (int n = 0; n < 4; ++n) {
            int col = colBase + n * 16 + laneM;
            if (col < N) {
                float bv = bias ? bias[col] : 0.f;
#pragma unroll
                for (int r = 0; r < 8; ++r) {
                    int row = rowBase + mi * 16 + hi * 8 + r;
                    size_t o = (size_t)row * N + col;
                    float val = acc[mi][n][r] + bv;
                    if (res)  val += res[o];
                    if (relu) val = fmaxf(val, 0.f);
                    if (outH) outH[o] = (f16)val;
                    else      outF[o] = val;
                }
            }
        }
    }
}

// ---------------------------------------------------------------------------
// Fused causal attention, online softmax. One block per (b, h); thread = query
// row t. K/V streamed through LDS in 32-key tiles. HD = 64.
// ---------------------------------------------------------------------------
__global__ __launch_bounds__(256) void attn_kernel(const float* __restrict__ Q,
                                                   const float* __restrict__ Km,
                                                   const float* __restrict__ Vm,
                                                   float* __restrict__ O) {
    const int bh = blockIdx.x;
    const int b  = bh / NHEAD;
    const int h  = bh % NHEAD;
    const int t  = threadIdx.x;

    __shared__ float Ks[32][HD];
    __shared__ float Vs[32][HD];

    const size_t rowq = (size_t)(b * SEQ + t) * DMODEL + h * HD;
    float q[HD], acc[HD];
#pragma unroll
    for (int d = 0; d < HD; ++d) { q[d] = Q[rowq + d]; acc[d] = 0.f; }

    float m = -1e30f, l = 0.f;
    const float scale = 0.125f;   // 64^-0.5

    for (int j0 = 0; j0 < SEQ; j0 += 32) {
#pragma unroll
        for (int r = 0; r < 8; ++r) {
            int lin = r * 256 + threadIdx.x;
            int jr = lin >> 6, c = lin & 63;
            size_t g = (size_t)(b * SEQ + j0 + jr) * DMODEL + h * HD + c;
            Ks[jr][c] = Km[g];
            Vs[jr][c] = Vm[g];
        }
        __syncthreads();

        int jmax = t - j0 + 1;                 // causal
        if (jmax > 32) jmax = 32;
        for (int j = 0; j < jmax; ++j) {
            float s = 0.f;
#pragma unroll
            for (int d = 0; d < HD; ++d) s += q[d] * Ks[j][d];
            s *= scale;
            float mn   = fmaxf(m, s);
            float corr = __expf(m - mn);
            float p    = __expf(s - mn);
            l = l * corr + p;
#pragma unroll
            for (int d = 0; d < HD; ++d) acc[d] = acc[d] * corr + p * Vs[j][d];
            m = mn;
        }
        __syncthreads();
    }

    float inv = 1.f / l;   // key 0 is always visible -> l > 0
    float* o = O + rowq;
#pragma unroll
    for (int d = 0; d < HD; ++d) o[d] = acc[d] * inv;
}

// ---------------------------------------------------------------------------
// Host orchestration
// ---------------------------------------------------------------------------
extern "C" void kernel_launch(void* const* d_in, const int* in_sizes, int n_in,
                              void* d_out, int out_size, void* d_ws, size_t ws_size,
                              hipStream_t stream) {
    const int*   idx    = (const int*)  d_in[0];
    const float* tok    = (const float*)d_in[1];
    const float* pos    = (const float*)d_in[2];
    const float* ln1_s  = (const float*)d_in[3];
    const float* ln1_b  = (const float*)d_in[4];
    const float* wq     = (const float*)d_in[5];
    const float* wk     = (const float*)d_in[6];
    const float* wv     = (const float*)d_in[7];
    const float* wo     = (const float*)d_in[8];
    const float* bo     = (const float*)d_in[9];
    const float* ln2_s  = (const float*)d_in[10];
    const float* ln2_b  = (const float*)d_in[11];
    const float* w1     = (const float*)d_in[12];
    const float* b1     = (const float*)d_in[13];
    const float* w2     = (const float*)d_in[14];
    const float* b2     = (const float*)d_in[15];
    const float* lnf_s  = (const float*)d_in[16];
    const float* lnf_b  = (const float*)d_in[17];
    const float* head_w = (const float*)d_in[18];
    const float* head_b = (const float*)d_in[19];
    float* logits = (float*)d_out;

    // ---- workspace carve (all chunk sizes are multiples of 32 bytes) ----
    char* p = (char*)d_ws;
    float* X  = (float*)p; p += (size_t)M * DMODEL * 4;   // residual stream
    float* Qb = (float*)p; p += (size_t)M * DMODEL * 4;
    float* Kb = (float*)p; p += (size_t)M * DMODEL * 4;
    float* Vb = (float*)p; p += (size_t)M * DMODEL * 4;
    float* Ab = (float*)p; p += (size_t)M * DMODEL * 4;   // attention out
    f16* H16  = (f16*)p;   p += (size_t)M * DMODEL * 2;   // GEMM input (f16)
    f16* F16  = (f16*)p;   p += (size_t)M * FF * 2;       // relu(ffn1) (f16)
    f16* Wc   = (f16*)p;                                   // converted weights

    const size_t DD  = (size_t)DMODEL * DMODEL;            // 147456
    const size_t DFF = (size_t)DMODEL * FF;                // 589824
    const size_t PL  = 4 * DD + 2 * DFF;                   // f16 elems / layer
    f16* headT = Wc + NLAYER * PL;                         // [VPAD, D]

    auto wconv = [&](const float* W, f16* out, int K, int N, int Npad) {
        size_t n = (size_t)Npad * K;
        wconv_kernel<<<(unsigned)((n + 255) / 256), 256, 0, stream>>>(W, out, K, N, Npad);
    };
    auto gemm = [&](const f16* Aa, const f16* Bt, const float* bias, const float* res,
                    float* oF, f16* oH, int K, int N, int Npad, int relu) {
        dim3 g(Npad / 64, M / 256);
        gemm_wmma_kernel<<<g, 256, 0, stream>>>(Aa, Bt, bias, res, oF, oH, K, N, relu);
    };

    // ---- convert all weights to transposed f16 (resident in L2 afterwards) ----
    for (int l = 0; l < NLAYER; ++l) {
        f16* base = Wc + l * PL;
        wconv(wq + l * DD,  base + 0 * DD, DMODEL, DMODEL, DMODEL);
        wconv(wk + l * DD,  base + 1 * DD, DMODEL, DMODEL, DMODEL);
        wconv(wv + l * DD,  base + 2 * DD, DMODEL, DMODEL, DMODEL);
        wconv(wo + l * DD,  base + 3 * DD, DMODEL, DMODEL, DMODEL);
        wconv(w1 + l * DFF, base + 4 * DD, DMODEL, FF, FF);           // [FF, D]
        wconv(w2 + l * DFF, base + 4 * DD + DFF, FF, DMODEL, DMODEL); // [D, FF]
    }
    wconv(head_w, headT, DMODEL, VOCAB, VPAD);

    // ---- embedding ----
    {
        size_t n = (size_t)M * DMODEL;
        embed_kernel<<<(unsigned)(n / 256), 256, 0, stream>>>(idx, tok, pos, X);
    }

    const unsigned lnGrid  = M / 8;
    const unsigned cvtGrid = (unsigned)((size_t)M * DMODEL / 256);

    // ---- transformer blocks ----
    for (int l = 0; l < NLAYER; ++l) {
        f16* base = Wc + l * PL;
        f16* wqT = base + 0 * DD;
        f16* wkT = base + 1 * DD;
        f16* wvT = base + 2 * DD;
        f16* woT = base + 3 * DD;
        f16* w1T = base + 4 * DD;
        f16* w2T = base + 4 * DD + DFF;

        ln_kernel<<<lnGrid, 256, 0, stream>>>(X, ln1_s + l * DMODEL, ln1_b + l * DMODEL, H16);
        gemm(H16, wqT, nullptr, nullptr, Qb, nullptr, DMODEL, DMODEL, DMODEL, 0);
        gemm(H16, wkT, nullptr, nullptr, Kb, nullptr, DMODEL, DMODEL, DMODEL, 0);
        gemm(H16, wvT, nullptr, nullptr, Vb, nullptr, DMODEL, DMODEL, DMODEL, 0);

        attn_kernel<<<BATCH * NHEAD, 256, 0, stream>>>(Qb, Kb, Vb, Ab);

        cvt_kernel<<<cvtGrid, 256, 0, stream>>>(Ab, H16);
        gemm(H16, woT, bo + l * DMODEL, X, X, nullptr, DMODEL, DMODEL, DMODEL, 0);

        ln_kernel<<<lnGrid, 256, 0, stream>>>(X, ln2_s + l * DMODEL, ln2_b + l * DMODEL, H16);
        gemm(H16, w1T, b1 + l * FF, nullptr, nullptr, F16, DMODEL, FF, FF, 1);  // +ReLU, f16
        gemm(F16, w2T, b2 + l * DMODEL, X, X, nullptr, FF, DMODEL, DMODEL, 0);
    }

    // ---- final LN + LM head ----
    ln_kernel<<<lnGrid, 256, 0, stream>>>(X, lnf_s, lnf_b, H16);
    gemm(H16, headT, head_b, nullptr, logits, nullptr, DMODEL, VOCAB, VPAD, 0);
}
